// SlotAttention_90649579749750
// MI455X (gfx1250) — compile-verified, matrix operand order checked by reading, not run on hardware
//
#include <hip/hip_runtime.h>

typedef __attribute__((ext_vector_type(16))) __bf16 v16bf;
typedef __attribute__((ext_vector_type(8)))  float  v8f;
typedef __attribute__((__vector_size__(16))) int    i32x4;

#define DEV __device__ __forceinline__
#define AS1 __attribute__((address_space(1)))
#define AS3 __attribute__((address_space(3)))

#if defined(__HIP_DEVICE_COMPILE__) && \
    __has_builtin(__builtin_amdgcn_global_load_async_to_lds_b128) && \
    __has_builtin(__builtin_amdgcn_s_wait_asynccnt)
#define USE_ASYNC_LDS 1
#else
#define USE_ASYNC_LDS 0
#endif

constexpr float EPS_    = 1e-8f;
constexpr float LN_EPS_ = 1e-5f;
constexpr float SCALE_  = 0.04419417382415922f; // 512^-0.5

// ---------------------------------------------------------------- reductions
DEV float block_sum(float v, float* red) {
  const int tid = threadIdx.x;
  red[tid] = v;
  __syncthreads();
  #pragma unroll
  for (int s = 128; s > 0; s >>= 1) {
    if (tid < s) red[tid] += red[tid + s];
    __syncthreads();
  }
  float r = red[0];
  __syncthreads();
  return r;
}

// ------------------------------------------------- prep: slots copy + W pack
__global__ void __launch_bounds__(256)
prep_kernel(const float* __restrict__ Wk, const float* __restrict__ Wv,
            const float* __restrict__ init_slots,
            unsigned short* __restrict__ wt_u, float* __restrict__ slots_g) {
  __bf16* Wt = (__bf16*)wt_u;
  const int idx = blockIdx.x * 256 + threadIdx.x;
  if (idx < 1024 * 512) {               // Wt[n][k] = W{k|v}[k][n], bf16, K-major
    const int n = idx >> 9;
    const int k = idx & 511;
    const float w = (n < 512) ? Wk[(size_t)k * 512 + n]
                              : Wv[(size_t)k * 512 + (n - 512)];
    Wt[(size_t)n * 512 + k] = (__bf16)w;
  }
  if (idx < 32 * 2 * 512) slots_g[idx] = init_slots[idx];
}

// --------------------------------------- layernorm(inputs) -> bf16 rows
__global__ void __launch_bounds__(256)
ln_in_kernel(const float* __restrict__ x, const float* __restrict__ g,
             const float* __restrict__ bta, unsigned short* __restrict__ xln_u) {
  const int wave = threadIdx.x >> 5;
  const int lane = threadIdx.x & 31;
  const size_t row = (size_t)blockIdx.x * 8 + wave;
  const float* xr = x + row * 512;
  float v[16];
  float s = 0.f, ss = 0.f;
  #pragma unroll
  for (int i = 0; i < 4; ++i) {
    const float4 t = ((const float4*)xr)[lane * 4 + i];
    v[4*i+0] = t.x; v[4*i+1] = t.y; v[4*i+2] = t.z; v[4*i+3] = t.w;
    s  += t.x + t.y + t.z + t.w;
    ss += t.x*t.x + t.y*t.y + t.z*t.z + t.w*t.w;
  }
  #pragma unroll
  for (int off = 16; off > 0; off >>= 1) {   // wave32 tree reduce
    s  += __shfl_xor(s,  off, 32);
    ss += __shfl_xor(ss, off, 32);
  }
  const float mean = s * (1.f / 512.f);
  const float var  = ss * (1.f / 512.f) - mean * mean;
  const float inv  = rsqrtf(var + LN_EPS_);
  v16bf frag;
  #pragma unroll
  for (int i = 0; i < 16; ++i) {
    const int j = lane * 16 + i;
    frag[i] = (__bf16)((v[i] - mean) * inv * g[j] + bta[j]);
  }
  *(v16bf*)(((__bf16*)xln_u) + row * 512 + lane * 16) = frag;
}

// ------------------------------- fused K/V projection GEMM via bf16 WMMA
// C[65536 x 1024] = xln[65536 x 512] * Wt^T ; cols 0..511 -> k(f32)+bk,
// cols 512..1023 -> v(bf16)+bv.  128x128 tile / WG (8 waves), B tile staged
// in LDS with double-buffered async global->LDS copies (ASYNCcnt).
__global__ void __launch_bounds__(256)
kv_wmma_kernel(const unsigned short* __restrict__ xln_u,
               const unsigned short* __restrict__ wt_u,
               const float* __restrict__ bk, const float* __restrict__ bv,
               float* __restrict__ k_out, unsigned short* __restrict__ v_out_u) {
  constexpr int KC   = 64;   // K-chunk depth (bf16 elements)
  constexpr int LSTR = 72;   // padded LDS row stride (144 B) to spread banks
  __shared__ __align__(32) unsigned short sB[2][128 * LSTR];  // ~36 KB

  const __bf16* xln  = (const __bf16*)xln_u;
  __bf16*       vout = (__bf16*)v_out_u;

  const int tid  = threadIdx.x;
  const int wave = tid >> 5;
  const int lane = tid & 31;
  const int half = lane >> 4;     // K-half selector (fragment layout)
  const int lm   = lane & 15;

  // grid.x = N-blocks (8): consecutive WGs share the same A tile -> L2 reuse
  const int    nbase = blockIdx.x * 128;
  const size_t mrow  = (size_t)blockIdx.y * 128 + wave * 16 + lm;

  // cooperative copy of one B chunk: 128 cols x 64 halves (16 KB), 4 x b128/thread
  auto issue_chunk = [&](int c) {
    unsigned short* dst = &sB[c & 1][0];
    const unsigned short* src = wt_u + (size_t)nbase * 512 + (size_t)c * KC;
    #pragma unroll
    for (int r = 0; r < 4; ++r) {
      const int vi  = tid + 256 * r;
      const int col = vi >> 3;          // 0..127
      const int ko  = (vi & 7) * 8;     // 0..56 halves (16B granules)
      const unsigned short* g = src + (size_t)col * 512 + ko;
      unsigned short*       l = dst + col * LSTR + ko;
#if USE_ASYNC_LDS
      __builtin_amdgcn_global_load_async_to_lds_b128(
          (AS1 i32x4*)g, (AS3 i32x4*)l, 0, 0);
#else
      *(uint4*)l = *(const uint4*)g;
#endif
    }
  };

  v8f acc[8];
  const v8f zero = {};
  #pragma unroll
  for (int t = 0; t < 8; ++t) acc[t] = zero;

  const __bf16* ap = xln + mrow * 512 + half * 16;

  issue_chunk(0);
#if !USE_ASYNC_LDS
  __syncthreads();
#endif

  #pragma unroll 1
  for (int c = 0; c < 8; ++c) {
#if USE_ASYNC_LDS
    if (c < 7) {                       // prefetch next chunk, keep 4 in flight
      issue_chunk(c + 1);
      __builtin_amdgcn_s_wait_asynccnt(4);
    } else {
      __builtin_amdgcn_s_wait_asynccnt(0);
    }
    __syncthreads();                   // chunk c visible to all waves
#endif
    const unsigned short* bbase = &sB[c & 1][0];
    #pragma unroll
    for (int kk = 0; kk < KC; kk += 32) {
      const v16bf af = *(const v16bf*)(ap + c * KC + kk);
      #pragma unroll
      for (int t = 0; t < 8; ++t) {
        const v16bf bf = *(const v16bf*)
            ((const __bf16*)(bbase + (t * 16 + lm) * LSTR + half * 16 + kk));
        acc[t] = __builtin_amdgcn_wmma_f32_16x16x32_bf16(
            /*neg_a=*/false, af, /*neg_b=*/false, bf,
            /*c_mod=*/(short)0, acc[t], /*reuse_a=*/false, /*reuse_b=*/false);
      }
    }
    __syncthreads();                   // done reading buf before it is rewritten
#if !USE_ASYNC_LDS
    if (c < 7) { issue_chunk(c + 1); __syncthreads(); }
#endif
  }

  // C/D layout: lane%16 = N, vgpr r -> M = r + 8*(lane/16)
  const size_t rbase = (size_t)blockIdx.y * 128 + wave * 16 + 8 * half;
  #pragma unroll
  for (int t = 0; t < 8; ++t) {
    const int col = nbase + t * 16 + lm;
    #pragma unroll
    for (int r = 0; r < 8; ++r) {
      const size_t row = rbase + r;
      const float  val = acc[t][r];
      if (col < 512) k_out[row * 512 + col] = val + bk[col];
      else           vout[row * 512 + (col - 512)] = (__bf16)(val + bv[col - 512]);
    }
  }
}

// ------------------------------- one slot-attention iteration (1 WG / batch)
__global__ void __launch_bounds__(256)
slot_iter_kernel(float* __restrict__ slots_g,             // [B,2,512] in/out
                 float* __restrict__ q_out,               // [B,2,512]
                 float* __restrict__ attn_out,            // [B,2,2048]
                 const float* __restrict__ k_f,           // [B,2048,512]
                 const unsigned short* __restrict__ v_bf_u,
                 const float* __restrict__ Wq, const float* __restrict__ bq,
                 const float* __restrict__ ln_s_g, const float* __restrict__ ln_s_b,
                 const float* __restrict__ wih, const float* __restrict__ whh,
                 const float* __restrict__ bih, const float* __restrict__ bhh,
                 const float* __restrict__ ln_f_g, const float* __restrict__ ln_f_b,
                 const float* __restrict__ W1, const float* __restrict__ b1,
                 const float* __restrict__ W2, const float* __restrict__ b2) {
  __shared__ float sh_h[1024];      // slots_prev
  __shared__ float sh_q[1024];
  __shared__ float sh_attn[4096];
  __shared__ float sh_upd[1024];
  __shared__ float sh_hnew[1024];
  __shared__ float sh_tmp[1024];    // s_ln then h_ln
  __shared__ float sh_h1[1024];
  __shared__ float red[256];

  const int tid = threadIdx.x;
  const int b   = blockIdx.x;
  const __bf16* v_bf = (const __bf16*)v_bf_u;

  for (int i = tid; i < 1024; i += 256) sh_h[i] = slots_g[b * 1024 + i];
  __syncthreads();

  // layernorm(slots) -> sh_tmp
  for (int i = 0; i < 2; ++i) {
    const float a = sh_h[i * 512 + tid], c = sh_h[i * 512 + 256 + tid];
    const float mean = block_sum(a + c, red) * (1.f / 512.f);
    const float d0 = a - mean, d1 = c - mean;
    const float var = block_sum(d0 * d0 + d1 * d1, red) * (1.f / 512.f);
    const float inv = rsqrtf(var + LN_EPS_);
    sh_tmp[i * 512 + tid]       = d0 * inv * ln_s_g[tid]       + ln_s_b[tid];
    sh_tmp[i * 512 + 256 + tid] = d1 * inv * ln_s_g[256 + tid] + ln_s_b[256 + tid];
  }
  __syncthreads();

  // q = s_ln @ Wq + bq
  for (int idx = tid; idx < 1024; idx += 256) {
    const int i = idx >> 9, j = idx & 511;
    float acc = bq[j];
    const float* s = sh_tmp + i * 512;
    #pragma unroll 4
    for (int d = 0; d < 512; ++d) acc += s[d] * Wq[(size_t)d * 512 + j];
    sh_q[idx] = acc;
    q_out[b * 1024 + idx] = acc;
  }
  __syncthreads();

  // dots + softmax over slot dim (S=2 -> pointwise per column)
  float p0 = 0.f, p1 = 0.f;
  for (int j = tid; j < 2048; j += 256) {
    const float* kr = k_f + ((size_t)(b * 2048 + j)) * 512;
    float a0 = 0.f, a1 = 0.f;
    #pragma unroll 4
    for (int d = 0; d < 512; ++d) { const float kv = kr[d]; a0 += sh_q[d] * kv; a1 += sh_q[512 + d] * kv; }
    a0 *= SCALE_; a1 *= SCALE_;
    const float m  = fmaxf(a0, a1);
    float e0 = expf(a0 - m), e1 = expf(a1 - m);
    const float inv = 1.f / (e0 + e1);
    e0 *= inv; e1 *= inv;
    sh_attn[j] = e0; sh_attn[2048 + j] = e1;
    p0 += e0; p1 += e1;
  }
  const float s0 = block_sum(p0, red);
  const float s1 = block_sum(p1, red);
  const float r0 = 1.f / (s0 + EPS_), r1 = 1.f / (s1 + EPS_);
  for (int j = tid; j < 2048; j += 256) {
    const float a0 = sh_attn[j] * r0, a1 = sh_attn[2048 + j] * r1;
    sh_attn[j] = a0; sh_attn[2048 + j] = a1;
    attn_out[b * 4096 + j] = a0; attn_out[b * 4096 + 2048 + j] = a1;
  }
  __syncthreads();

  // updates[i][d] = sum_j v[b,j,d] * attn[i][j]
  for (int idx = tid; idx < 1024; idx += 256) {
    const int i = idx >> 9, d = idx & 511;
    const __bf16* vp = v_bf + (size_t)b * 2048 * 512 + d;
    const float*  at = sh_attn + i * 2048;
    float acc = 0.f;
    #pragma unroll 4
    for (int j = 0; j < 2048; ++j) acc += (float)vp[(size_t)j * 512] * at[j];
    sh_upd[idx] = acc;
  }
  __syncthreads();

  // GRU cell (gate order r,z,n) fully fused per (row,d)
  for (int idx = tid; idx < 1024; idx += 256) {
    const int i = idx >> 9, d = idx & 511;
    const float* u = sh_upd + i * 512;
    const float* h = sh_h   + i * 512;
    float xr = bih[d], xz = bih[512 + d], xn = bih[1024 + d];
    float hr = bhh[d], hz = bhh[512 + d], hn = bhh[1024 + d];
    const float* wr = wih + (size_t)d * 512;
    const float* wz = wih + (size_t)(512 + d) * 512;
    const float* wn = wih + (size_t)(1024 + d) * 512;
    const float* vr = whh + (size_t)d * 512;
    const float* vz = whh + (size_t)(512 + d) * 512;
    const float* vn = whh + (size_t)(1024 + d) * 512;
    #pragma unroll 2
    for (int e = 0; e < 512; ++e) {
      const float ue = u[e], he = h[e];
      xr += ue * wr[e]; xz += ue * wz[e]; xn += ue * wn[e];
      hr += he * vr[e]; hz += he * vz[e]; hn += he * vn[e];
    }
    const float r = 1.f / (1.f + expf(-(xr + hr)));
    const float z = 1.f / (1.f + expf(-(xz + hz)));
    const float n = tanhf(xn + r * hn);
    sh_hnew[idx] = (1.f - z) * n + z * h[d];
  }
  __syncthreads();

  // layernorm(h_new) -> sh_tmp
  for (int i = 0; i < 2; ++i) {
    const float a = sh_hnew[i * 512 + tid], c = sh_hnew[i * 512 + 256 + tid];
    const float mean = block_sum(a + c, red) * (1.f / 512.f);
    const float d0 = a - mean, d1 = c - mean;
    const float var = block_sum(d0 * d0 + d1 * d1, red) * (1.f / 512.f);
    const float inv = rsqrtf(var + LN_EPS_);
    sh_tmp[i * 512 + tid]       = d0 * inv * ln_f_g[tid]       + ln_f_b[tid];
    sh_tmp[i * 512 + 256 + tid] = d1 * inv * ln_f_g[256 + tid] + ln_f_b[256 + tid];
  }
  __syncthreads();

  // MLP layer 1 (relu)
  for (int idx = tid; idx < 1024; idx += 256) {
    const int i = idx >> 9, j = idx & 511;
    float acc = b1[j];
    const float* hl = sh_tmp + i * 512;
    #pragma unroll 4
    for (int d = 0; d < 512; ++d) acc += hl[d] * W1[(size_t)d * 512 + j];
    sh_h1[idx] = fmaxf(acc, 0.f);
  }
  __syncthreads();

  // MLP layer 2 + residual -> slots
  for (int idx = tid; idx < 1024; idx += 256) {
    const int i = idx >> 9, j = idx & 511;
    float acc = b2[j];
    const float* hh = sh_h1 + i * 512;
    #pragma unroll 4
    for (int d = 0; d < 512; ++d) acc += hh[d] * W2[(size_t)d * 512 + j];
    slots_g[b * 1024 + idx] = sh_hnew[idx] + acc;
  }
}

// --------------------------------------------------------------------- host
extern "C" void kernel_launch(void* const* d_in, const int* in_sizes, int n_in,
                              void* d_out, int out_size, void* d_ws, size_t ws_size,
                              hipStream_t stream) {
  const float* inputs  = (const float*)d_in[0];
  const float* init_s  = (const float*)d_in[1];
  const float* Wq      = (const float*)d_in[2];
  const float* bq      = (const float*)d_in[3];
  const float* Wk      = (const float*)d_in[4];
  const float* bk      = (const float*)d_in[5];
  const float* Wv      = (const float*)d_in[6];
  const float* bv      = (const float*)d_in[7];
  const float* ln_in_g = (const float*)d_in[8];
  const float* ln_in_b = (const float*)d_in[9];
  const float* ln_s_g  = (const float*)d_in[10];
  const float* ln_s_b  = (const float*)d_in[11];
  const float* ln_f_g  = (const float*)d_in[12];
  const float* ln_f_b  = (const float*)d_in[13];
  const float* wih     = (const float*)d_in[14];
  const float* whh     = (const float*)d_in[15];
  const float* bih     = (const float*)d_in[16];
  const float* bhh     = (const float*)d_in[17];
  const float* W1      = (const float*)d_in[18];
  const float* b1      = (const float*)d_in[19];
  const float* W2      = (const float*)d_in[20];
  const float* b2      = (const float*)d_in[21];

  char* ws = (char*)d_ws;
  unsigned short* xln = (unsigned short*)ws;                        // 64 MiB
  unsigned short* Wt  = (unsigned short*)(ws + (size_t)67108864);   //  1 MiB
  unsigned short* vbf = (unsigned short*)(ws + (size_t)68157440);   // 64 MiB

  float* out      = (float*)d_out;
  float* slots_g  = out;                       // [32,2,512]
  float* q_out    = out + 32768;               // [32,2,512]
  float* k_out    = out + 65536;               // [32,2048,512]
  float* attn_out = out + 65536 + 33554432;    // [32,2,2048]

  prep_kernel <<<2048, 256, 0, stream>>>(Wk, Wv, init_s, Wt, slots_g);
  ln_in_kernel<<<8192, 256, 0, stream>>>(inputs, ln_in_g, ln_in_b, xln);
  dim3 grid(8, 512);   // x = N-blocks (fastest) so A tiles are reused in L2
  kv_wmma_kernel<<<grid, 256, 0, stream>>>(xln, Wt, bk, bv, k_out, vbf);
  for (int it = 0; it < 3; ++it) {
    slot_iter_kernel<<<32, 256, 0, stream>>>(slots_g, q_out, attn_out, k_out, vbf,
        Wq, bq, ln_s_g, ln_s_b, wih, whh, bih, bhh,
        ln_f_g, ln_f_b, W1, b1, W2, b2);
  }
}
